// FullAttnNode_7043746365630
// MI455X (gfx1250) — compile-verified, matrix operand order checked by reading, not run on hardware
//
#include <hip/hip_runtime.h>

#define D_MODEL 2048
#define T_SEQ   2048
#define N_BATCH 2
#define N_HEADS 16
#define HEAD_DIM 128
#define NTOK (N_BATCH * T_SEQ)   // 4096

typedef __attribute__((ext_vector_type(16))) __bf16 v16bf;
typedef __attribute__((ext_vector_type(8)))  __bf16 v8bf;
typedef __attribute__((ext_vector_type(8)))  float  v8f;
typedef int v4i_vs __attribute__((vector_size(16)));   // 16-byte chunk for async DMA

#define AS1 __attribute__((address_space(1)))
#define AS3 __attribute__((address_space(3)))

#if __has_builtin(__builtin_amdgcn_global_load_async_to_lds_b128) && \
    __has_builtin(__builtin_amdgcn_s_wait_asynccnt)
#define HAVE_ASYNC 1
#else
#define HAVE_ASYNC 0
#endif

// 16-byte global -> LDS copy; async (no VGPR roundtrip, ASYNCcnt) when available.
static __device__ __forceinline__ void copy16_g2l(const __bf16* g, __bf16* l) {
#if HAVE_ASYNC
  __builtin_amdgcn_global_load_async_to_lds_b128((AS1 v4i_vs*)g, (AS3 v4i_vs*)l, 0, 0);
#else
  *(v8bf*)l = *(const v8bf*)g;
#endif
}

static __device__ __forceinline__ void async_wait() {
#if HAVE_ASYNC
  __builtin_amdgcn_s_wait_asynccnt(0);
#endif
}

static __device__ __forceinline__ v8f wmma_bf16(v16bf a, v16bf b, v8f c) {
  // 8 args: (neg_a, A, neg_b, B, c_mod, C, reuse_a, reuse_b)
  return __builtin_amdgcn_wmma_f32_16x16x32_bf16(false, a, false, b, (short)0, c,
                                                 false, false);
}

// ---------------- fp32 -> bf16 weight conversion ----------------
__global__ __launch_bounds__(256) void k_f32_to_bf16(const float* __restrict__ in,
                                                     __bf16* __restrict__ out, int n) {
  int i = blockIdx.x * 1024 + threadIdx.x * 4;
  if (i + 3 < n) {
    float4 v = *(const float4*)(in + i);
    out[i + 0] = (__bf16)v.x;
    out[i + 1] = (__bf16)v.y;
    out[i + 2] = (__bf16)v.z;
    out[i + 3] = (__bf16)v.w;
  }
}

// ---------------- RMSNorm fused with bf16 convert ----------------
__global__ __launch_bounds__(256) void k_rmsnorm(const float* __restrict__ x,
                                                 const float* __restrict__ w,
                                                 __bf16* __restrict__ xn) {
  const int row = blockIdx.x;
  const float* xr = x + (size_t)row * D_MODEL;
  float ss = 0.f;
  for (int i = threadIdx.x; i < D_MODEL; i += 256) { float v = xr[i]; ss += v * v; }
  #pragma unroll
  for (int d = 16; d >= 1; d >>= 1) ss += __shfl_xor(ss, d);
  __shared__ float red[8];
  const int wave = threadIdx.x >> 5, lane = threadIdx.x & 31;
  if (lane == 0) red[wave] = ss;
  __syncthreads();
  if (threadIdx.x == 0) {
    float s = 0.f;
    #pragma unroll
    for (int i2 = 0; i2 < 8; i2++) s += red[i2];
    red[0] = rsqrtf(s / (float)D_MODEL + 1e-6f);
  }
  __syncthreads();
  const float inv = red[0];
  for (int i = threadIdx.x; i < D_MODEL; i += 256)
    xn[(size_t)row * D_MODEL + i] = (__bf16)(xr[i] * inv * w[i]);
}

// ---------------- GEMM: C[M,N] = A[M,K] @ W[N,K]^T + bias ----------------
// K = N = D_MODEL, A/W bf16 row-major. Block tile 128x128, 8 waves of 32x64.
// Double-buffered LDS staging: async loads for tile k+1 overlap WMMAs on tile k.
// mode 0: bf16 scatter -> [B,H,T,HD]   (Q,K)
// mode 1: bf16 scatter -> [B,H,HD,T]   (V transposed)
// mode 2: f32 -> [M,N]                 (final projection)
__global__ __launch_bounds__(256) void k_gemm(const __bf16* __restrict__ A,
                                              const __bf16* __restrict__ W,
                                              const float*  __restrict__ bias,
                                              __bf16* __restrict__ out_bf,
                                              float*  __restrict__ out_f,
                                              int mode) {
  __shared__ __attribute__((aligned(16))) __bf16 lA[2][128 * 32];
  __shared__ __attribute__((aligned(16))) __bf16 lB[2][128 * 32];
  const int m0 = blockIdx.y * 128;
  const int n0 = blockIdx.x * 128;
  const int tid = threadIdx.x;
  const int wave = tid >> 5, lane = tid & 31;
  const int wy = wave >> 1, wx = wave & 1;   // 4x2 wave grid
  const int r = lane & 15, hi = lane >> 4;

  v8f acc[2][4] = {};

  // stage one 128x32 tile pair into buffer `buf` (16B chunks, 2 per thread each)
  #define STAGE_GEMM(buf, kk)                                                     \
    do {                                                                          \
      _Pragma("unroll")                                                           \
      for (int c_ = 0; c_ < 2; c_++) {                                            \
        int idx_ = tid + c_ * 256;                                                \
        int row_ = idx_ >> 2, sub_ = idx_ & 3;                                    \
        copy16_g2l(&A[(size_t)(m0 + row_) * D_MODEL + (kk) + sub_ * 8],           \
                   &lA[buf][row_ * 32 + sub_ * 8]);                               \
        copy16_g2l(&W[(size_t)(n0 + row_) * D_MODEL + (kk) + sub_ * 8],           \
                   &lB[buf][row_ * 32 + sub_ * 8]);                               \
      }                                                                           \
    } while (0)

  STAGE_GEMM(0, 0);
  async_wait();
  __syncthreads();

  const int KT = D_MODEL / 32;
  for (int kt = 0; kt < KT; kt++) {
    const int cur = kt & 1;
    if (kt + 1 < KT) STAGE_GEMM(cur ^ 1, (kt + 1) * 32);

    // fragment gather per CDNA5 16-bit A/B layout: lane=row, K halves split by hi
    v16bf af[2], bfr[4];
    #pragma unroll
    for (int i = 0; i < 2; i++) {
      const __bf16* p = &lA[cur][(wy * 32 + i * 16 + r) * 32];
      ((v8bf*)&af[i])[0] = *(const v8bf*)(p + hi * 8);
      ((v8bf*)&af[i])[1] = *(const v8bf*)(p + 16 + hi * 8);
    }
    #pragma unroll
    for (int j = 0; j < 4; j++) {
      const __bf16* p = &lB[cur][(wx * 64 + j * 16 + r) * 32];
      ((v8bf*)&bfr[j])[0] = *(const v8bf*)(p + hi * 8);
      ((v8bf*)&bfr[j])[1] = *(const v8bf*)(p + 16 + hi * 8);
    }
    #pragma unroll
    for (int i = 0; i < 2; i++)
      #pragma unroll
      for (int j = 0; j < 4; j++)
        acc[i][j] = wmma_bf16(af[i], bfr[j], acc[i][j]);

    async_wait();     // this wave's loads for buffer cur^1 landed
    __syncthreads();  // everyone done reading cur + everyone's loads visible
  }
  #undef STAGE_GEMM

  // epilogue: C layout -> lane holds col r, rows vr + 8*hi
  #pragma unroll
  for (int i = 0; i < 2; i++) {
    #pragma unroll
    for (int j = 0; j < 4; j++) {
      const int ng = n0 + wx * 64 + j * 16 + r;
      const float bv = bias[ng];
      #pragma unroll
      for (int vr = 0; vr < 8; vr++) {
        const int m = m0 + wy * 32 + i * 16 + vr + 8 * hi;
        const float val = acc[i][j][vr] + bv;
        if (mode == 2) {
          out_f[(size_t)m * D_MODEL + ng] = val;
        } else {
          const int bi = m >> 11, t = m & (T_SEQ - 1);
          const int h = ng >> 7, hd = ng & (HEAD_DIM - 1);
          if (mode == 0)
            out_bf[(((size_t)(bi * N_HEADS + h) * T_SEQ) + t) * HEAD_DIM + hd] = (__bf16)val;
          else
            out_bf[(((size_t)(bi * N_HEADS + h) * HEAD_DIM) + hd) * T_SEQ + t] = (__bf16)val;
        }
      }
    }
  }
}

// ---------------- causal flash attention ----------------
// grid (T/128, B*H); block 256 = 8 waves; wave owns 16 query rows.
// K and V^T blocks staged once per key-block into LDS (shared by all 8 waves).
__global__ __launch_bounds__(256) void k_attn(const __bf16* __restrict__ Q,
                                              const __bf16* __restrict__ K,
                                              const __bf16* __restrict__ Vt,
                                              __bf16* __restrict__ O) {
  const int qb = blockIdx.x;
  const int bh = blockIdx.y;
  const __bf16* Qh = Q  + (size_t)bh * T_SEQ * HEAD_DIM;
  const __bf16* Kh = K  + (size_t)bh * T_SEQ * HEAD_DIM;
  const __bf16* Vh = Vt + (size_t)bh * HEAD_DIM * T_SEQ;
  const int tid = threadIdx.x;
  const int wave = tid >> 5, lane = tid & 31;
  const int r = lane & 15, hi = lane >> 4;
  const int qrow0 = qb * 128 + wave * 16;

  __shared__ __attribute__((aligned(16))) __bf16 lK[128 * 128];    // 32 KB
  __shared__ __attribute__((aligned(16))) __bf16 lV[128 * 128];    // 32 KB
  __shared__ __attribute__((aligned(16))) __bf16 lP[8][16 * 128];  // 32 KB
  __bf16* lp = lP[wave];

  const float scale = 0.088388347648318447f;  // 1/sqrt(128)

  // Q fragments pinned in registers for the whole pass (4 K-chunks of 32)
  v16bf qf[4];
  #pragma unroll
  for (int c = 0; c < 4; c++) {
    const __bf16* p = &Qh[(size_t)(qrow0 + r) * HEAD_DIM + c * 32];
    ((v8bf*)&qf[c])[0] = *(const v8bf*)(p + hi * 8);
    ((v8bf*)&qf[c])[1] = *(const v8bf*)(p + 16 + hi * 8);
  }

  float rowmax[8], rowsum[8];
  v8f oacc[8] = {};
  #pragma unroll
  for (int vr = 0; vr < 8; vr++) { rowmax[vr] = -__builtin_inff(); rowsum[vr] = 0.f; }

  for (int j = 0; j <= qb; j++) {
    const int k0 = j * 128;

    __syncthreads();  // all waves done reading lK/lV from previous iteration
    // stage K block [key 0..127][d 0..127] and V^T block [d 0..127][key 0..127]
    #pragma unroll
    for (int i = 0; i < 8; i++) {
      const int c = tid + i * 256;            // 0..2047 16B chunks
      const int row = c >> 4, sub = c & 15;
      copy16_g2l(&Kh[(size_t)(k0 + row) * HEAD_DIM + sub * 8], &lK[row * 128 + sub * 8]);
      copy16_g2l(&Vh[(size_t)row * T_SEQ + k0 + sub * 8],      &lV[row * 128 + sub * 8]);
    }
    async_wait();
    __syncthreads();

    // S = Q @ K^T : 8 key tiles x 4 K-chunks = 32 WMMAs (B operand = rows of K)
    v8f s[8];
    #pragma unroll
    for (int t = 0; t < 8; t++) {
      v8f cacc = {};
      #pragma unroll
      for (int c = 0; c < 4; c++) {
        v16bf kf;
        const __bf16* p = &lK[(t * 16 + r) * 128 + c * 32];
        ((v8bf*)&kf)[0] = *(const v8bf*)(p + hi * 8);
        ((v8bf*)&kf)[1] = *(const v8bf*)(p + 16 + hi * 8);
        cacc = wmma_bf16(qf[c], kf, cacc);
      }
      s[t] = cacc;
    }
    // online softmax: row lives across 16 lanes of a half-wave (xor masks 1..8)
    #pragma unroll
    for (int vr = 0; vr < 8; vr++) {
      const int m = qrow0 + vr + 8 * hi;
      float mx = -__builtin_inff();
      #pragma unroll
      for (int t = 0; t < 8; t++) {
        const int kg = k0 + t * 16 + r;
        float v = s[t][vr] * scale;
        v = (kg <= m) ? v : -__builtin_inff();
        s[t][vr] = v;
        mx = fmaxf(mx, v);
      }
      #pragma unroll
      for (int d = 1; d < 16; d <<= 1) mx = fmaxf(mx, __shfl_xor(mx, d));
      const float nm = fmaxf(rowmax[vr], mx);
      const float alpha = __expf(rowmax[vr] - nm);
      float rs = 0.f;
      #pragma unroll
      for (int t = 0; t < 8; t++) {
        const float e = __expf(s[t][vr] - nm);
        s[t][vr] = e;
        rs += e;
      }
      #pragma unroll
      for (int d = 1; d < 16; d <<= 1) rs += __shfl_xor(rs, d);
      rowsum[vr] = rowsum[vr] * alpha + rs;
      rowmax[vr] = nm;
      #pragma unroll
      for (int t = 0; t < 8; t++) oacc[t][vr] *= alpha;
    }
    // re-layout P (C-layout -> A-layout) through the wave-private LDS slab
    #pragma unroll
    for (int t = 0; t < 8; t++)
      #pragma unroll
      for (int vr = 0; vr < 8; vr++)
        lp[(vr + 8 * hi) * 128 + t * 16 + r] = (__bf16)s[t][vr];

    // O += P @ V  (B operand = rows of V^T from LDS)
    #pragma unroll
    for (int kc = 0; kc < 4; kc++) {
      v16bf pf;
      const __bf16* pp = &lp[r * 128 + kc * 32];
      ((v8bf*)&pf)[0] = *(const v8bf*)(pp + hi * 8);
      ((v8bf*)&pf)[1] = *(const v8bf*)(pp + 16 + hi * 8);
      #pragma unroll
      for (int dt = 0; dt < 8; dt++) {
        v16bf vf;
        const __bf16* vp = &lV[(dt * 16 + r) * 128 + kc * 32];
        ((v8bf*)&vf)[0] = *(const v8bf*)(vp + hi * 8);
        ((v8bf*)&vf)[1] = *(const v8bf*)(vp + 16 + hi * 8);
        oacc[dt] = wmma_bf16(pf, vf, oacc[dt]);
      }
    }
  }

  const int b = bh >> 4, h = bh & (N_HEADS - 1);
  #pragma unroll
  for (int dt = 0; dt < 8; dt++) {
    #pragma unroll
    for (int vr = 0; vr < 8; vr++) {
      const int t = qrow0 + vr + 8 * hi;
      const int d = dt * 16 + r;
      const float v = oacc[dt][vr] / rowsum[vr];
      O[((size_t)(b * T_SEQ + t)) * D_MODEL + h * HEAD_DIM + d] = (__bf16)v;
    }
  }
}

// ---------------- launch ----------------
extern "C" void kernel_launch(void* const* d_in, const int* in_sizes, int n_in,
                              void* d_out, int out_size, void* d_ws, size_t ws_size,
                              hipStream_t stream) {
  (void)in_sizes; (void)n_in; (void)out_size; (void)ws_size;
  const float* x      = (const float*)d_in[0];
  const float* norm_w = (const float*)d_in[1];
  const float* wq = (const float*)d_in[2]; const float* bq = (const float*)d_in[3];
  const float* wk = (const float*)d_in[4]; const float* bk = (const float*)d_in[5];
  const float* wv = (const float*)d_in[6]; const float* bv = (const float*)d_in[7];
  const float* wo = (const float*)d_in[8]; const float* bo = (const float*)d_in[9];
  float* out = (float*)d_out;

  char* ws = (char*)d_ws;
  const size_t ACT = (size_t)NTOK * D_MODEL * 2;      // 16 MB bf16 activations
  const size_t WSZ = (size_t)D_MODEL * D_MODEL * 2;   // 8 MB bf16 weights
  __bf16* xn  = (__bf16*)(ws + 0 * ACT);
  __bf16* qb_ = (__bf16*)(ws + 1 * ACT);
  __bf16* kb_ = (__bf16*)(ws + 2 * ACT);
  __bf16* vtb = (__bf16*)(ws + 3 * ACT);
  __bf16* ao  = (__bf16*)(ws + 4 * ACT);
  __bf16* wqb = (__bf16*)(ws + 5 * ACT + 0 * WSZ);
  __bf16* wkb = (__bf16*)(ws + 5 * ACT + 1 * WSZ);
  __bf16* wvb = (__bf16*)(ws + 5 * ACT + 2 * WSZ);
  __bf16* wob = (__bf16*)(ws + 5 * ACT + 3 * WSZ);

  const int nW = D_MODEL * D_MODEL;
  k_f32_to_bf16<<<dim3(nW / 1024), dim3(256), 0, stream>>>(wq, wqb, nW);
  k_f32_to_bf16<<<dim3(nW / 1024), dim3(256), 0, stream>>>(wk, wkb, nW);
  k_f32_to_bf16<<<dim3(nW / 1024), dim3(256), 0, stream>>>(wv, wvb, nW);
  k_f32_to_bf16<<<dim3(nW / 1024), dim3(256), 0, stream>>>(wo, wob, nW);

  k_rmsnorm<<<dim3(NTOK), dim3(256), 0, stream>>>(x, norm_w, xn);

  dim3 gg(D_MODEL / 128, NTOK / 128);   // (16, 32)
  k_gemm<<<gg, dim3(256), 0, stream>>>(xn, wqb, bq, qb_, nullptr, 0);
  k_gemm<<<gg, dim3(256), 0, stream>>>(xn, wkb, bk, kb_, nullptr, 0);
  k_gemm<<<gg, dim3(256), 0, stream>>>(xn, wvb, bv, vtb, nullptr, 1);

  k_attn<<<dim3(T_SEQ / 128, N_BATCH * N_HEADS), dim3(256), 0, stream>>>(qb_, kb_, vtb, ao);

  k_gemm<<<gg, dim3(256), 0, stream>>>(ao, wob, bo, nullptr, out, 2);
}
